// Attention_60533269070530
// MI455X (gfx1250) — matrix operand order repaired and checked by reading, not compiled, not run on hardware
//
#include <hip/hip_runtime.h>
#include <hip/hip_bf16.h>

typedef __attribute__((ext_vector_type(8)))  __bf16 v8bf;
typedef __attribute__((ext_vector_type(16))) __bf16 v16bf;
typedef __attribute__((ext_vector_type(8)))  float  v8f;
typedef int vi2 __attribute__((vector_size(2 * sizeof(int))));

#define AS1_ __attribute__((address_space(1)))
#define AS3_ __attribute__((address_space(3)))

#define B_    2
#define T_    2048
#define D_    2048
#define N_    16
#define KV_   4
#define H_    256
#define NH_   4096   /* N_*H_ */
#define KH_   1024   /* KV_*H_ */
#define QKVC  6144   /* NH_ + 2*KH_ */
#define WINDOW_ 1024
#define SOFTCAP 50.0f

// Async global->LDS path (gfx1250), guarded so either toolchain compiles.
#if defined(__has_builtin)
#  if __has_builtin(__builtin_amdgcn_global_load_async_to_lds_b64) && \
      __has_builtin(__builtin_amdgcn_s_wait_asynccnt)
#    define ASYNC_LDS 1
#  endif
#endif
#ifndef ASYNC_LDS
#  define ASYNC_LDS 0
#endif

static __device__ __forceinline__ v8bf ldv8(const __bf16* p) {
    return *(const v8bf*)p;
}
static __device__ __forceinline__ v16bf mk16(v8bf lo, v8bf hi) {
    union { v16bf v; v8bf h[2]; } u;
    u.h[0] = lo; u.h[1] = hi;
    return u.v;
}
static __device__ __forceinline__ v8f wmma_bf16(v16bf a, v16bf b, v8f c) {
    // (neg_a, A, neg_b, B, c_mod, C, reuse_a, reuse_b)
    return __builtin_amdgcn_wmma_f32_16x16x32_bf16(false, a, false, b, (short)0, c, false, false);
}

// Copy 8 bytes global -> LDS (one lane's share of an A k-slice).
static __device__ __forceinline__ void copy8_g2l(const __bf16* g, __bf16* l) {
#if ASYNC_LDS
    vi2* gg = (vi2*)(uintptr_t)g;   // generic, const stripped
    vi2* ll = (vi2*)l;              // generic (LDS object)
    __builtin_amdgcn_global_load_async_to_lds_b64(
        (AS1_ vi2*)gg, (AS3_ vi2*)ll, 0, 0);
#else
    *(uint2*)l = *(const uint2*)g;
#endif
}

// ---------------------------------------------------------------------------
// Prep kernels: fp32 -> bf16 conversions + weight transposes (K-dim contiguous)
// ---------------------------------------------------------------------------
__global__ void k_cvt_x(const float* __restrict__ x, __bf16* __restrict__ xb) {
    int i = blockIdx.x * 256 + threadIdx.x;
    if (i < B_ * T_ * D_) xb[i] = (__bf16)x[i];
}

// wt[c][d] = w_q[n][d][h],  c = n*H+h  (rows 0..4095 of combined QKV weight)
__global__ void k_build_wq(const float* __restrict__ wq, __bf16* __restrict__ wt) {
    int i = blockIdx.x * 256 + threadIdx.x;          // over NH_*D_
    if (i >= NH_ * D_) return;
    int d = i % D_;
    int c = i / D_;
    int n = c >> 8, h = c & 255;
    wt[(size_t)i] = (__bf16)wq[((size_t)n * D_ + d) * H_ + h];
}

// rows 4096..6143: wt[NH_ + e*KH_ + kk*H + h][d] = w_kv[e][kk][d][h]
__global__ void k_build_wkv(const float* __restrict__ wkv, __bf16* __restrict__ wt) {
    int i = blockIdx.x * 256 + threadIdx.x;          // over 2*KV_*H_*D_
    if (i >= 2 * KV_ * H_ * D_) return;
    int d = i % D_;
    int c = i / D_;                                   // c in [0, 2048)
    int e  = c >> 10;
    int kk = (c >> 8) & 3;
    int h  = c & 255;
    wt[((size_t)(NH_ + c)) * D_ + d] =
        (__bf16)wkv[(((size_t)(e * KV_ + kk)) * D_ + d) * H_ + h];
}

// wot[d][c] = w_o[n][h][d] = w_o_flat[c*D + d]
__global__ void k_build_wo(const float* __restrict__ wo, __bf16* __restrict__ wot) {
    int i = blockIdx.x * 256 + threadIdx.x;          // over D_*NH_
    if (i >= D_ * NH_) return;
    int c = i % NH_;
    int d = i / NH_;
    wot[(size_t)i] = (__bf16)wo[(size_t)c * D_ + d];
}

// ---------------------------------------------------------------------------
// Unified WMMA GEMM: C(16 x 256 per block) = A(16 x KDIM) * Bt(cols x KDIM)^T.
// 4 waves/block, wave w owns 16x64.  The A tile is shared by all 4 waves:
// staged through double-buffered LDS slices with async global->LDS copies
// (ASYNCcnt), while B fragments are double-buffered in VGPRs so WMMA overlaps
// the loads instead of hitting s_wait_loadcnt 0 every issue.
// MODE 0: QKV epilogue (q row-major / k row-major / v transposed), bf16 out.
// MODE 1: fp32 output, ld = D_.
// ---------------------------------------------------------------------------
template <int KDIM, int MODE>
__global__ __launch_bounds__(128) void k_gemm(const __bf16* __restrict__ A,
                                              const __bf16* __restrict__ Bt,
                                              __bf16* __restrict__ qb,
                                              __bf16* __restrict__ kb,
                                              __bf16* __restrict__ vt,
                                              float* __restrict__ fout) {
    __shared__ __bf16 Atile[2][16 * 32];

    const int tid  = threadIdx.x;
    const int lane = tid & 31;
    const int wave = tid >> 5;
    const int l    = lane & 15;
    const int half = lane >> 4;
    const int m0 = blockIdx.y * 16;
    const int n0 = blockIdx.x * 256 + wave * 64;

    // Per-thread share of the cooperative A-slice copy: 8 bytes each.
    const __bf16* aSrc = A + (size_t)(m0 + (tid >> 3)) * KDIM + (tid & 7) * 4;
    __bf16*       aDst0 = &Atile[0][tid * 4];
    __bf16*       aDst1 = &Atile[1][tid * 4];

    const __bf16* brow = Bt + (size_t)(n0 + l) * KDIM + half * 16;

    v8f acc[4];
    #pragma unroll
    for (int j = 0; j < 4; ++j) acc[j] = (v8f){0,0,0,0,0,0,0,0};

    const int NK = KDIM / 32;

    // Prologue: stage A slice 0, load B fragments for k0 = 0.
    copy8_g2l(aSrc, aDst0);
    v16bf bcur[4], bnext[4];
    #pragma unroll
    for (int j = 0; j < 4; ++j)
        bcur[j] = mk16(ldv8(brow + (size_t)j * 16 * KDIM),
                       ldv8(brow + (size_t)j * 16 * KDIM + 8));

    for (int kk = 0; kk < NK; ++kk) {
        const int s = kk & 1;
        const bool hasnext = (kk + 1) < NK;
        if (hasnext) {
            // Stage next A slice (buffer s^1 is free: end-of-iter barrier).
            copy8_g2l(aSrc + (kk + 1) * 32, s ? aDst0 : aDst1);
            const int k0n = (kk + 1) * 32;
            #pragma unroll
            for (int j = 0; j < 4; ++j)
                bnext[j] = mk16(ldv8(brow + (size_t)j * 16 * KDIM + k0n),
                                ldv8(brow + (size_t)j * 16 * KDIM + k0n + 8));
            __builtin_prefetch(aSrc + (kk + 2) * 32, 0, 1);
        }
#if ASYNC_LDS
        if (hasnext) __builtin_amdgcn_s_wait_asynccnt(1);  // slice s done
        else         __builtin_amdgcn_s_wait_asynccnt(0);
#endif
        __syncthreads();   // all waves' slice-s copies visible

        const __bf16* as = &Atile[s][l * 32 + half * 16];
        v16bf a = mk16(*(const v8bf*)as, *(const v8bf*)(as + 8));
        #pragma unroll
        for (int j = 0; j < 4; ++j) acc[j] = wmma_bf16(a, bcur[j], acc[j]);

        if (hasnext) {
            #pragma unroll
            for (int j = 0; j < 4; ++j) bcur[j] = bnext[j];
        }
        __syncthreads();   // done reading slice s before it is overwritten
    }

    #pragma unroll
    for (int j = 0; j < 4; ++j) {
        int col = n0 + 16 * j + l;
        #pragma unroll
        for (int r = 0; r < 8; ++r) {
            int row = m0 + r + half * 8;                  // global token (b*T + t)
            float val = acc[j][r];
            if (MODE == 1) {
                fout[(size_t)row * D_ + col] = val;
            } else if (col < NH_) {
                qb[(size_t)row * NH_ + col] = (__bf16)val;
            } else if (col < NH_ + KH_) {
                kb[(size_t)row * KH_ + (col - NH_)] = (__bf16)val;
            } else {
                int vc = col - (NH_ + KH_);
                int bb2 = row >> 11, tt = row & (T_ - 1);
                vt[((size_t)(bb2 * KV_ + (vc >> 8)) * H_ + (vc & 255)) * T_ + tt] =
                    (__bf16)val;
            }
        }
    }
}

// ---------------------------------------------------------------------------
// RoPE in-place on qb (with 1/sqrt(H) scale) and kb.
// ---------------------------------------------------------------------------
__global__ void k_rope(__bf16* __restrict__ qb, __bf16* __restrict__ kb,
                       const int* __restrict__ segpos) {
    int idx = blockIdx.x * 256 + threadIdx.x;
    const int total = B_ * T_ * (N_ + KV_) * (H_ / 2);
    if (idx >= total) return;
    int i    = idx & 127;
    int rest = idx >> 7;
    int head = rest % (N_ + KV_);
    int tok  = rest / (N_ + KV_);

    float pos = (float)segpos[tok];
    float inv_ts = __expf(-(float)i * (9.210340371976184f / 128.0f)); // 10000^(-i/128)
    float ang = pos * inv_ts;
    float s, c;
    __sincosf(ang, &s, &c);

    __bf16* p;
    float scale;
    if (head < N_) { p = qb + ((size_t)tok * N_ + head) * H_;         scale = 0.0625f; }
    else           { p = kb + ((size_t)tok * KV_ + (head - N_)) * H_; scale = 1.0f; }

    float f = (float)p[i];
    float g = (float)p[i + 128];
    p[i]       = (__bf16)((f * c - g * s) * scale);
    p[i + 128] = (__bf16)((g * c + f * s) * scale);
}

// ---------------------------------------------------------------------------
// Sliding-window GQA attention, one wave = 16 queries x 1 head.
// S^T = K*Q^T so probs land directly in the A-fragment layout for P*V.
// Fixed softmax max = SOFTCAP (logits tanh-capped), so no online rescale.
// K and V fragment streams are double-buffered in VGPRs.
// ---------------------------------------------------------------------------
__global__ __launch_bounds__(32) void k_attn(const __bf16* __restrict__ qb,
                                             const __bf16* __restrict__ kb,
                                             const __bf16* __restrict__ vt,
                                             __bf16* __restrict__ enc) {
    const int lane = threadIdx.x & 31;
    const int l    = lane & 15;
    const int half = lane >> 4;
    const int q0 = blockIdx.x * 16;
    const int n  = blockIdx.y;
    const int b  = blockIdx.z;
    const int kv = n >> 2;

    const __bf16* Q = qb + ((size_t)(b * T_ + q0) * N_ + n) * H_;  // +t*N_*H_
    const __bf16* K = kb + ((size_t)(b * T_) * KV_ + kv) * H_;     // +s*KV_*H_
    const __bf16* V = vt + (size_t)(b * KV_ + kv) * H_ * T_;       // +h*T_
    const int t_lane = q0 + l;

    // Preload Q fragments (B operand of S^T): lane l = query q0+l,
    // k-dim = h: lanes<16 -> k0..k0+7,k0+16..k0+23, lanes>=16 -> k0+8..k0+15,k0+24..k0+31.
    v16bf qf[8];
    #pragma unroll
    for (int kk = 0; kk < 8; ++kk) {
        const __bf16* qp = Q + (size_t)l * (N_ * H_) + kk * 32 + half * 8;
        qf[kk] = mk16(ldv8(qp), ldv8(qp + 16));
    }

    v8f o[16];
    #pragma unroll
    for (int j = 0; j < 16; ++j) o[j] = (v8f){0,0,0,0,0,0,0,0};
    float lsum = 0.0f;

    int sstart = q0 - (WINDOW_ - 1);
    if (sstart < 0) sstart = 0;
    sstart &= ~31;
    const int send = q0 + 15;

    for (int s0 = sstart; s0 <= send; s0 += 32) {
        const __bf16* kbase = K + (size_t)(s0 + l) * (KV_ * H_) + half * 8;

        // ---- S^T tiles: tile0 = keys s0..s0+15, tile1 = keys s0+16..s0+31
        v8f st0 = (v8f){0,0,0,0,0,0,0,0};
        v8f st1 = (v8f){0,0,0,0,0,0,0,0};
        v16bf a0c = mk16(ldv8(kbase), ldv8(kbase + 16));
        v16bf a1c = mk16(ldv8(kbase + (size_t)16 * (KV_ * H_)),
                         ldv8(kbase + (size_t)16 * (KV_ * H_) + 16));
        #pragma unroll
        for (int kk = 0; kk < 8; ++kk) {
            v16bf a0n, a1n;
            if (kk < 7) {
                const __bf16* ka = kbase + (kk + 1) * 32;
                a0n = mk16(ldv8(ka), ldv8(ka + 16));
                a1n = mk16(ldv8(ka + (size_t)16 * (KV_ * H_)),
                           ldv8(ka + (size_t)16 * (KV_ * H_) + 16));
            }
            st0 = wmma_bf16(a0c, qf[kk], st0);
            st1 = wmma_bf16(a1c, qf[kk], st1);
            a0c = a0n;
            a1c = a1n;
        }

        // ---- softcap + mask + exp(l - cap); pack straight into PV A-fragment
        v16bf p;
        #pragma unroll
        for (int r = 0; r < 8; ++r) {
            int s_key0 = s0 + r + half * 8;
            int s_key1 = s_key0 + 16;
            float x0 = SOFTCAP * tanhf(st0[r] * (1.0f / SOFTCAP));
            float x1 = SOFTCAP * tanhf(st1[r] * (1.0f / SOFTCAP));
            bool v0 = (s_key0 <= t_lane) && (s_key0 > t_lane - WINDOW_);
            bool v1 = (s_key1 <= t_lane) && (s_key1 > t_lane - WINDOW_);
            float p0 = v0 ? __expf(x0 - SOFTCAP) : 0.0f;
            float p1 = v1 ? __expf(x1 - SOFTCAP) : 0.0f;
            lsum += p0 + p1;
            p[r]     = (__bf16)p0;
            p[8 + r] = (__bf16)p1;
        }

        // ---- O += P * V (B-fragment: lane = h column, k-dim = keys, from vT)
        const __bf16* vbase = V + (size_t)l * T_ + s0 + half * 8;
        v16bf bvc = mk16(ldv8(vbase), ldv8(vbase + 16));
        #pragma unroll
        for (int j = 0; j < 16; ++j) {
            v16bf bvn;
            if (j < 15) {
                const __bf16* vp = vbase + (size_t)(j + 1) * 16 * T_;
                bvn = mk16(ldv8(vp), ldv8(vp + 16));
            }
            o[j] = wmma_bf16(p, bvc, o[j]);
            bvc = bvn;
        }
    }

    // lanes l and l+16 both hold query q0+l partial sums
    lsum += __shfl_xor(lsum, 16, 32);
    float inv = 1.0f / lsum;

    // O rows: VGPR r <-> query r + half*8; its sum lives at lane (r + half*8)
    float sc[8];
    #pragma unroll
    for (int r = 0; r < 8; ++r) sc[r] = __shfl(inv, r + half * 8, 32);

    #pragma unroll
    for (int j = 0; j < 16; ++j) {
        #pragma unroll
        for (int r = 0; r < 8; ++r) {
            size_t row = (size_t)(b * T_ + q0 + r + half * 8);
            enc[(row * N_ + n) * H_ + j * 16 + l] = (__bf16)(o[j][r] * sc[r]);
        }
    }
}

// ---------------------------------------------------------------------------
extern "C" void kernel_launch(void* const* d_in, const int* in_sizes, int n_in,
                              void* d_out, int out_size, void* d_ws, size_t ws_size,
                              hipStream_t stream) {
    (void)in_sizes; (void)n_in; (void)out_size; (void)ws_size;
    const float* x      = (const float*)d_in[0];
    const int*   segpos = (const int*)d_in[1];
    // d_in[2] = attn_mask (causal tril) — reconstructed analytically, unused
    const float* wq  = (const float*)d_in[3];
    const float* wkv = (const float*)d_in[4];
    const float* wo  = (const float*)d_in[5];
    float* out = (float*)d_out;

    char* ws = (char*)d_ws;
    size_t off = 0;
    auto alloc = [&](size_t bytes) -> char* {
        char* p = ws + off;
        off += (bytes + 255) & ~(size_t)255;
        return p;
    };
    __bf16* xb   = (__bf16*)alloc((size_t)B_ * T_ * D_ * 2);        // 16.8 MB
    __bf16* wt   = (__bf16*)alloc((size_t)QKVC * D_ * 2);           // 25.2 MB
    __bf16* wot  = (__bf16*)alloc((size_t)D_ * NH_ * 2);            // 16.8 MB
    __bf16* qb   = (__bf16*)alloc((size_t)B_ * T_ * NH_ * 2);       // 33.6 MB
    __bf16* kbuf = (__bf16*)alloc((size_t)B_ * T_ * KH_ * 2);       //  8.4 MB
    __bf16* vt   = (__bf16*)alloc((size_t)B_ * KV_ * H_ * T_ * 2);  //  8.4 MB
    __bf16* enc  = (__bf16*)alloc((size_t)B_ * T_ * NH_ * 2);       // 33.6 MB

    // 1) bf16 conversions / weight transposes
    k_cvt_x    <<<(B_ * T_ * D_ + 255) / 256,       256, 0, stream>>>(x, xb);
    k_build_wq <<<(NH_ * D_ + 255) / 256,           256, 0, stream>>>(wq, wt);
    k_build_wkv<<<(2 * KV_ * H_ * D_ + 255) / 256,  256, 0, stream>>>(wkv, wt);
    k_build_wo <<<(D_ * NH_ + 255) / 256,           256, 0, stream>>>(wo, wot);

    // 2) fused QKV projection (WMMA bf16, async-LDS staged A), v transposed
    k_gemm<D_, 0><<<dim3(QKVC / 256, (B_ * T_) / 16, 1), 128, 0, stream>>>(
        xb, wt, qb, kbuf, vt, nullptr);

    // 3) RoPE (+ q scale) in place
    k_rope<<<(B_ * T_ * (N_ + KV_) * (H_ / 2) + 255) / 256, 256, 0, stream>>>(
        qb, kbuf, segpos);

    // 4) sliding-window GQA attention (WMMA bf16, softcap softmax)
    k_attn<<<dim3(T_ / 16, N_, B_), 32, 0, stream>>>(qb, kbuf, vt, enc);

    // 5) output projection (WMMA bf16 -> fp32)
    k_gemm<NH_, 1><<<dim3(D_ / 256, (B_ * T_) / 16, 1), 128, 0, stream>>>(
        enc, wot, nullptr, nullptr, nullptr, out);
}
